// SALAD_44289702756913
// MI455X (gfx1250) — matrix operand order repaired and checked by reading, not compiled, not run on hardware
//
#include <hip/hip_runtime.h>
#include <hip/hip_bf16.h>
#include <math.h>

// ---------------- CDNA5 WMMA types ----------------
typedef __attribute__((ext_vector_type(8)))  float   v8f;
typedef __attribute__((ext_vector_type(16))) __bf16  v16bf;
typedef int v4i_t __attribute__((vector_size(16)));   // matches async-LDS builtin param

union Frag { v16bf v; unsigned int u[8]; };

#define B_   32
#define N_   1024      // N-1 feature rows
#define D_   768
#define K_   64
#define CD_  128
#define TD_  256
#define NCAT 192       // CD_ + K_
#define XROW 787200    // 1025*768, fp32 elements per batch of x
#define FOUT 8448      // TD_ + K_*CD_

#if __has_builtin(__builtin_amdgcn_global_load_async_to_lds_b128)
#define HAS_ASYNC 1
#else
#define HAS_ASYNC 0
#endif

// 16-byte global -> LDS copy: async DMA (ASYNCcnt) when available, else direct.
__device__ __forceinline__ void cp16(const void* g, void* l) {
#if HAS_ASYNC
  v4i_t* gp = (v4i_t*)g;   // drop const, generic AS
  v4i_t* lp = (v4i_t*)l;
  __builtin_amdgcn_global_load_async_to_lds_b128(
      (__attribute__((address_space(1))) v4i_t*)gp,
      (__attribute__((address_space(3))) v4i_t*)lp,
      0, 0);
#else
  *(uint4*)l = *(const uint4*)g;
#endif
}
__device__ __forceinline__ void wait_async() {
#if HAS_ASYNC
#if __has_builtin(__builtin_amdgcn_s_wait_asynccnt)
  __builtin_amdgcn_s_wait_asynccnt(0);
#else
  asm volatile("s_wait_asynccnt 0" ::: "memory");
#endif
#endif
}

__device__ __forceinline__ unsigned int pkbf(float a, float b) {
  unsigned short lo = __builtin_bit_cast(unsigned short, (__bf16)a);
  unsigned short hi = __builtin_bit_cast(unsigned short, (__bf16)b);
  return (unsigned int)lo | ((unsigned int)hi << 16);
}
__device__ __forceinline__ unsigned short f2bf(float a) {
  return __builtin_bit_cast(unsigned short, (__bf16)a);
}
// A-matrix 16x32 bf16 fragment: VGPR v holds packed K-pair index j (K=2j,2j+1)
// lanes 0-15: K 0..7 (v0..3), 16..23 (v4..7); lanes 16-31: K 8..15, 24..31.
__device__ __forceinline__ int a_jmap(int v, int hi) {
  return (v & 3) + (hi << 2) + ((v >> 2) << 3);
}

// ---------------- 1) repack weights to bf16, N-major ----------------
__global__ void k_convw(const float* __restrict__ Wf, const float* __restrict__ Ws,
                        const float* __restrict__ Wt,
                        unsigned short* __restrict__ wcat, unsigned short* __restrict__ wt) {
  int e = blockIdx.x * blockDim.x + threadIdx.x;
  const int tot1 = NCAT * D_;
  if (e < tot1) {
    int c = e / D_, k = e % D_;
    float v = (c < CD_) ? Wf[k * CD_ + c] : Ws[k * K_ + (c - CD_)];
    wcat[c * D_ + k] = f2bf(v);
  } else {
    int e2 = e - tot1;
    if (e2 < TD_ * D_) {
      int c = e2 / D_, k = e2 % D_;
      wt[c * D_ + k] = f2bf(Wt[k * TD_ + c]);
    }
  }
}

// ---------------- 2) f,s = fi @ [Wf|Ws]  (M=32768,N=192,K=768) ----------------
// Double-buffered LDS tiles filled by async global->LDS DMA; bf16 WMMA compute.
__global__ void __launch_bounds__(256)
k_gemm_fs(const float* __restrict__ x, const float* __restrict__ bf,
          const float* __restrict__ bs, const float* __restrict__ sharp,
          const unsigned int* __restrict__ wcat_u,
          unsigned short* __restrict__ fbt, float* __restrict__ sm) {
  __shared__ float        Af[2][128 * 32];   // A tile, raw fp32 (2 x 16KB)
  __shared__ unsigned int Bu[2][NCAT * 16];  // B tile, bf16 pairs (2 x 12KB)
  const int t = threadIdx.x, w = t >> 5, lane = t & 31;
  const int hi = lane >> 4, ln = lane & 15;
  const int b = blockIdx.x >> 3;                  // 8 blocks per batch
  const int nbase = (blockIdx.x & 7) * 128;       // row-tile within batch
  const float* fiB = x + (size_t)b * XROW + D_;   // skip token row 0

  auto stage = [&](int buf, int k0) {
    #pragma unroll
    for (int i = 0; i < 4; i++) {                 // A: 1024 chunks of 16B
      const int idx = t + (i << 8);
      const int r = idx >> 3, cj = idx & 7;       // 8 chunks per 32-float row
      cp16(fiB + (size_t)(nbase + r) * D_ + k0 + (cj << 2), &Af[buf][idx << 2]);
    }
    #pragma unroll
    for (int i = 0; i < 3; i++) {                 // B: 768 chunks of 16B
      const int idx = t + (i << 8);
      const int c = idx >> 2, jj = idx & 3;       // 4 chunks per 16-uint col slice
      cp16(wcat_u + c * (D_ / 2) + (k0 >> 1) + (jj << 2), &Bu[buf][idx << 2]);
    }
  };

  v8f acc[12];
  #pragma unroll
  for (int i = 0; i < 12; i++) acc[i] = (v8f){0.f,0.f,0.f,0.f,0.f,0.f,0.f,0.f};

  stage(0, 0);
  wait_async();
  __syncthreads();
  for (int ks = 0; ks < D_ / 32; ks++) {
    const int cur = ks & 1;
    if (ks + 1 < D_ / 32) stage(cur ^ 1, (ks + 1) << 5);  // DMA next tile during WMMAs
    Frag a;
    const float* af = &Af[cur][((w << 4) + ln) << 5];
    #pragma unroll
    for (int v = 0; v < 8; v++) {                 // fp32 pair -> packed bf16
      const int j = a_jmap(v, hi);
      const float2 p = *(const float2*)(af + 2 * j);
      a.u[v] = pkbf(p.x, p.y);
    }
    #pragma unroll
    for (int nt = 0; nt < 12; nt++) {
      Frag bb;
      const unsigned int* bp = &Bu[cur][((nt << 4) + ln) << 4];
      #pragma unroll
      for (int v = 0; v < 8; v++) bb.u[v] = bp[a_jmap(v, hi)];
      acc[nt] = __builtin_amdgcn_wmma_f32_16x16x32_bf16(false, a.v, false, bb.v,
                                                        (short)0, acc[nt], false, false);
    }
    wait_async();
    __syncthreads();
  }
  // Epilogue. C/D layout: lane -> col (N), VGPR v -> row M = v + hi*8 (consecutive).
  const float sscale = sharp[0] * 10.0f;          // sharpness / REG, REG = 0.1
  const int n0 = nbase + (w << 4) + (hi << 3);    // 8 consecutive n per lane
  #pragma unroll
  for (int nt = 0; nt < 12; nt++) {
    const int c = (nt << 4) + ln;
    if (c < CD_) {                                // f part -> bf16, transposed [c][n]
      const float bias = bf[c];
      unsigned int pk[4];
      #pragma unroll
      for (int j = 0; j < 4; j++)
        pk[j] = pkbf(acc[nt][2 * j] + bias, acc[nt][2 * j + 1] + bias);
      *(uint4*)(fbt + (((size_t)(b * CD_ + c)) << 10) + n0) =
          make_uint4(pk[0], pk[1], pk[2], pk[3]);
    } else {                                      // s part -> fp32 logits, [k][n]
      const int ks = c - CD_;
      const float bias = bs[ks];
      float* dst = sm + (((size_t)(b * K_ + ks)) << 10) + n0;
      *(float4*)(dst) = make_float4((acc[nt][0] + bias) * sscale, (acc[nt][1] + bias) * sscale,
                                    (acc[nt][2] + bias) * sscale, (acc[nt][3] + bias) * sscale);
      *(float4*)(dst + 4) = make_float4((acc[nt][4] + bias) * sscale, (acc[nt][5] + bias) * sscale,
                                        (acc[nt][6] + bias) * sscale, (acc[nt][7] + bias) * sscale);
    }
  }
}

// ---------------- 3) t = t_global @ Wt  (32 x 768 x 256) ----------------
__global__ void __launch_bounds__(256)
k_gemm_t(const float* __restrict__ x, const float* __restrict__ bt,
         const unsigned int* __restrict__ wt_u, float* __restrict__ out) {
  const int t = threadIdx.x, w = t >> 5, lane = t & 31;
  const int hi = lane >> 4, ln = lane & 15;
  const int tile = blockIdx.x * 8 + w;            // 0..31 = 2 Mtiles x 16 Ntiles
  const int mt = tile >> 4, nt = tile & 15;
  v8f acc = (v8f){0.f,0.f,0.f,0.f,0.f,0.f,0.f,0.f};
  const int arow = (mt << 4) + ln;                // batch index (M dim)
  const unsigned int* bp = wt_u + (size_t)((nt << 4) + ln) * (D_ / 2);
  for (int k0 = 0; k0 < D_; k0 += 32) {
    Frag a, bb;
    #pragma unroll
    for (int v = 0; v < 8; v++) {
      const int j = a_jmap(v, hi);
      const float2 p = *(const float2*)(x + (size_t)arow * XROW + k0 + 2 * j);
      a.u[v] = pkbf(p.x, p.y);
      bb.u[v] = bp[(k0 >> 1) + a_jmap(v, hi)];
    }
    acc = __builtin_amdgcn_wmma_f32_16x16x32_bf16(false, a.v, false, bb.v,
                                                  (short)0, acc, false, false);
  }
  const int col = (nt << 4) + ln;
  #pragma unroll
  for (int v = 0; v < 8; v++) {
    const int row = (mt << 4) + v + (hi << 3);
    out[(size_t)row * FOUT + col] = acc[v] + bt[col];
  }
}

// ---------------- 4) Sinkhorn (log-domain OT), one workgroup per batch ----------------
__global__ void __launch_bounds__(1024)
k_sinkhorn(const float* __restrict__ sm, const float* __restrict__ dust,
           unsigned short* __restrict__ pbf, float* __restrict__ rsum) {
  __shared__ float uu[K_ + 1];
  __shared__ float vv[N_];
  const int b = blockIdx.x, t = threadIdx.x, w = t >> 5, lane = t & 31;
  const float* S = sm + (((size_t)b * K_) << 10);
  const float Mdust = dust[0] * 10.0f;            // dust_bin / REG
  const float LOG_A = -4.174387269895637f;        // -log(K+1) = -log 65
  const float LOG_B = -6.931471805599453f;        // -log(N-1) = -log 1024
  if (t <= K_) uu[t] = 0.f;
  vv[t] = 0.f;
  __syncthreads();

  for (int it = 0; it < 5; it++) {
    // u-update: wave w owns rows w, w+32; wave 0 also the dust row (k=64).
    for (int pass = 0; pass < 3; pass++) {
      int k;
      if (pass == 0) k = w;
      else if (pass == 1) k = w + 32;
      else { if (w != 0) break; k = K_; }
      const float* rowp = S + ((size_t)k << 10);
      float m = -__builtin_inff(), s = 0.f;
      for (int i = 0; i < 32; i++) {              // online logsumexp, 32 elems/lane
        const int n = lane + (i << 5);
        float base;
        if (k < K_) base = rowp[n]; else base = Mdust;
        const float val = base + vv[n];
        if (val > m) { s = s * __expf(m - val) + 1.f; m = val; }
        else s += __expf(val - m);
      }
      for (int off = 16; off > 0; off >>= 1) {    // wave32 pairwise LSE merge
        const float mo = __shfl_xor(m, off, 32);
        const float so = __shfl_xor(s, off, 32);
        if (mo > m) { s = s * __expf(m - mo) + so; m = mo; }
        else s += so * __expf(mo - m);
      }
      if (lane == 0) uu[k] = LOG_A - (m + __logf(s));
    }
    __syncthreads();
    // v-update: thread t owns column n = t (coalesced column reads of S).
    {
      const float* colp = S + t;
      float m = -__builtin_inff(), s = 0.f;
      for (int k = 0; k < K_; k++) {
        const float val = colp[(size_t)k << 10] + uu[k];
        if (val > m) { s = s * __expf(m - val) + 1.f; m = val; }
        else s += __expf(val - m);
      }
      { const float val = Mdust + uu[K_];
        if (val > m) { s = s * __expf(m - val) + 1.f; m = val; }
        else s += __expf(val - m); }
      vv[t] = LOG_B - (m + __logf(s));
    }
    __syncthreads();
  }
  // p = exp(M+u+v) rows 0..63 -> bf16 (GEMM-3 A operand) + row sums.
  for (int pass = 0; pass < 2; pass++) {
    const int k = w + (pass << 5);
    const float* rowp = S + ((size_t)k << 10);
    unsigned short* pp = pbf + (((size_t)b * K_ + k) << 10);
    const float uk = uu[k];
    float sum = 0.f;
    for (int i = 0; i < 32; i++) {
      const int n = lane + (i << 5);
      const float pv = __expf(rowp[n] + uk + vv[n]);
      pp[n] = f2bf(pv);
      sum += pv;
    }
    for (int off = 16; off > 0; off >>= 1) sum += __shfl_xor(sum, off, 32);
    if (lane == 0) rsum[b * K_ + k] = sum;
  }
}

// ---------------- 5) v_agg = 2*p@f - rowsum*anchors  (per batch 64x1024x128) -------
__global__ void __launch_bounds__(256)
k_gemm_v(const unsigned int* __restrict__ p_u, const unsigned int* __restrict__ f_u,
         const float* __restrict__ rsum, const float* __restrict__ anchors,
         float* __restrict__ out) {
  const int b = blockIdx.x, t = threadIdx.x, w = t >> 5, lane = t & 31;
  const int hi = lane >> 4, ln = lane & 15;
  const int mt = w & 3, ntb = (w >> 2) << 2;      // 4 Mtiles x 2 groups of 4 Ntiles
  v8f acc[4];
  #pragma unroll
  for (int i = 0; i < 4; i++) acc[i] = (v8f){0.f,0.f,0.f,0.f,0.f,0.f,0.f,0.f};
  const unsigned int* ap = p_u + (((size_t)b * K_ + (mt << 4) + ln) << 9);
  for (int k0 = 0; k0 < N_; k0 += 32) {
    Frag a;
    #pragma unroll
    for (int v = 0; v < 8; v++) a.u[v] = ap[(k0 >> 1) + a_jmap(v, hi)];
    #pragma unroll
    for (int j = 0; j < 4; j++) {
      const int cc = ((ntb + j) << 4) + ln;
      const unsigned int* bp = f_u + (((size_t)b * CD_ + cc) << 9);
      Frag bb;
      #pragma unroll
      for (int v = 0; v < 8; v++) bb.u[v] = bp[(k0 >> 1) + a_jmap(v, hi)];
      acc[j] = __builtin_amdgcn_wmma_f32_16x16x32_bf16(false, a.v, false, bb.v,
                                                       (short)0, acc[j], false, false);
    }
  }
  #pragma unroll
  for (int j = 0; j < 4; j++) {
    const int c = ((ntb + j) << 4) + ln;
    #pragma unroll
    for (int v = 0; v < 8; v++) {
      const int kk = (mt << 4) + v + (hi << 3);
      const float val = 2.f * acc[j][v] - rsum[b * K_ + kk] * anchors[kk * CD_ + c];
      out[(size_t)b * FOUT + TD_ + kk * CD_ + c] = val;
    }
  }
}

// ---------------- 6) in-place L2 normalize per batch row ----------------
__global__ void __launch_bounds__(256)
k_norm(float* __restrict__ out) {
  __shared__ float red[8];
  const int b = blockIdx.x, t = threadIdx.x, w = t >> 5, lane = t & 31;
  float* p = out + (size_t)b * FOUT;
  float ss = 0.f;
  for (int i = t; i < FOUT; i += 256) { const float v = p[i]; ss += v * v; }
  for (int off = 16; off > 0; off >>= 1) ss += __shfl_xor(ss, off, 32);
  if (lane == 0) red[w] = ss;
  __syncthreads();
  if (t == 0) {
    float s = 0.f;
    for (int i = 0; i < 8; i++) s += red[i];
    red[0] = 1.0f / fmaxf(sqrtf(s), 1e-12f);
  }
  __syncthreads();
  const float inv = red[0];
  for (int i = t; i < FOUT; i += 256) p[i] *= inv;
}

// ---------------- host launcher ----------------
extern "C" void kernel_launch(void* const* d_in, const int* in_sizes, int n_in,
                              void* d_out, int out_size, void* d_ws, size_t ws_size,
                              hipStream_t stream) {
  (void)in_sizes; (void)n_in; (void)out_size; (void)ws_size;
  const float* x       = (const float*)d_in[0];
  const float* Wf      = (const float*)d_in[1];
  const float* bfp     = (const float*)d_in[2];
  const float* Ws      = (const float*)d_in[3];
  const float* bsp     = (const float*)d_in[4];
  const float* Wt      = (const float*)d_in[5];
  const float* btp     = (const float*)d_in[6];
  const float* anchors = (const float*)d_in[7];
  const float* dust    = (const float*)d_in[8];
  const float* sharp   = (const float*)d_in[9];
  float* out = (float*)d_out;

  char* ws = (char*)d_ws;
  size_t off = 0;
  auto take = [&](size_t bytes) {
    char* p = ws + off;
    off = (off + bytes + 255) & ~(size_t)255;
    return p;
  };
  unsigned short* wcat = (unsigned short*)take((size_t)NCAT * D_ * 2);
  unsigned short* wt   = (unsigned short*)take((size_t)TD_ * D_ * 2);
  unsigned short* fbt  = (unsigned short*)take((size_t)B_ * CD_ * N_ * 2); // f^T bf16
  float* sm            = (float*)take((size_t)B_ * K_ * N_ * 4);           // scaled logits
  unsigned short* pbf  = (unsigned short*)take((size_t)B_ * K_ * N_ * 2);  // p bf16
  float* rsum          = (float*)take((size_t)B_ * K_ * 4);

  k_convw<<<((NCAT + TD_) * D_ + 255) / 256, 256, 0, stream>>>(Wf, Ws, Wt, wcat, wt);
  k_gemm_fs<<<256, 256, 0, stream>>>(x, bfp, bsp, sharp,
                                     (const unsigned int*)wcat, fbt, sm);
  k_gemm_t<<<4, 256, 0, stream>>>(x, btp, (const unsigned int*)wt, out);
  k_sinkhorn<<<B_, 1024, 0, stream>>>(sm, dust, pbf, rsum);
  k_gemm_v<<<B_, 256, 0, stream>>>((const unsigned int*)pbf, (const unsigned int*)fbt,
                                   rsum, anchors, out);
  k_norm<<<B_, 256, 0, stream>>>(out);
}
